// MSALayer_16793322127902
// MI455X (gfx1250) — matrix operand order repaired and checked by reading, not compiled, not run on hardware
//
#include <hip/hip_runtime.h>
#include <hip/hip_bf16.h>
#include <math.h>

// MSA layer (Boltz) for gfx1250 / MI455X.
// Heavy contractions on v_wmma_f32_16x16x32_bf16 (fp32 accum). Operand
// storage layouts are chosen so the K dimension is contiguous: every WMMA
// fragment is assembled from 8 aligned 32-bit loads (bf16 K-pairs), not
// 16 scalar u16 loads. LN/softmax/SwiGLU stay on VALU (bandwidth-trivial).

#define DEV __device__ __forceinline__

typedef __bf16 bf16_t;
typedef bf16_t  v16bf __attribute__((ext_vector_type(16)));
typedef float   v8f   __attribute__((ext_vector_type(8)));
typedef unsigned v8u  __attribute__((ext_vector_type(8)));

constexpr int S  = 512;     // MSA depth
constexpr int NT = 384;     // tokens
constexpr int CM = 64;      // msa dim
constexpr int CZ = 128;     // pair dim
constexpr int H  = 8;       // heads
constexpr int DH = 32;      // head dim
constexpr int CP = 32;      // outer-product hidden
constexpr int HD = H * DH;  // 256
constexpr float EPS = 1e-5f;

DEV bf16_t f2bf(float x) {
  unsigned u = __builtin_bit_cast(unsigned, x);
  unsigned r = u + 0x7FFFu + ((u >> 16) & 1u);   // round-to-nearest-even
  unsigned short hs = (unsigned short)(r >> 16);
  return __builtin_bit_cast(bf16_t, hs);
}
DEV float bf2f(bf16_t b) {
  unsigned short hs = __builtin_bit_cast(unsigned short, b);
  unsigned u = ((unsigned)hs) << 16;
  return __builtin_bit_cast(float, u);
}
DEV float sigmoidf_(float x) { return 1.f / (1.f + expf(-x)); }

DEV v8f wmma_bf16(v16bf a, v16bf b, v8f c) {
  return __builtin_amdgcn_wmma_f32_16x16x32_bf16(false, a, false, b, (short)0, c,
                                                 false, false);
}

// Fragment index helper: dword r of an A-fragment holds K-pair
//   k(r) = akb + 2r,  akb = (lane>=16)*16
// Dword r of a B-fragment holds K-pair kb(r) = bkb + 2r, bkb = (lane>=16)*16.
DEV int a_koff(int akb, int r) { return akb + 2 * r; }

// ---------------------------------------------------------------------------
// K0: weight prep. Transposed bf16 copies so B-fragments are K-contiguous:
//   wo_pwa_t[n][k] (64x256),  wo_opm_t[n][k] (128x1024)
// ---------------------------------------------------------------------------
__global__ __launch_bounds__(256)
void k0_cvt(const float* __restrict__ pwa_wo, const float* __restrict__ opm_wo,
            bf16_t* __restrict__ wo_pwa_t, bf16_t* __restrict__ wo_opm_t) {
  int idx = blockIdx.x * 256 + threadIdx.x;
  if (idx < CM * HD) {                 // pwa_wo: [256][64] -> t[64][256]
    int n = idx / HD, k = idx % HD;
    wo_pwa_t[idx] = f2bf(pwa_wo[k * CM + n]);
  }
  if (idx < CZ * CP * CP) {            // opm_wo: [1024][128] -> t[128][1024]
    int n = idx / (CP * CP), k = idx % (CP * CP);
    wo_opm_t[idx] = f2bf(opm_wo[k * CZ + n]);
  }
}

// ---------------------------------------------------------------------------
// K1a: bias logits = LN(z) @ wz  -> logits[h][i][j]
// ---------------------------------------------------------------------------
__global__ __launch_bounds__(256)
void k1a_bias(const float* __restrict__ z0, const float* __restrict__ zw,
              const float* __restrict__ zb, const float* __restrict__ wz,
              float* __restrict__ logits) {
  int idx = blockIdx.x * 256 + threadIdx.x;
  if (idx >= NT * NT) return;
  const float* zr = z0 + (size_t)idx * CZ;
  float s1 = 0.f, s2 = 0.f;
  for (int k = 0; k < CZ; ++k) { float v = zr[k]; s1 += v; s2 += v * v; }
  float mu  = s1 * (1.f / CZ);
  float inv = rsqrtf(fmaxf(s2 * (1.f / CZ) - mu * mu, 0.f) + EPS);
  float acc[H];
  for (int h = 0; h < H; ++h) acc[h] = 0.f;
  for (int k = 0; k < CZ; ++k) {
    float zn = (zr[k] - mu) * inv * zw[k] + zb[k];
#pragma unroll
    for (int h = 0; h < H; ++h) acc[h] = fmaf(zn, wz[k * H + h], acc[h]);
  }
  for (int h = 0; h < H; ++h) logits[(size_t)h * NT * NT + idx] = acc[h];
}

// ---------------------------------------------------------------------------
// K1b: softmax over keys j for each (h,i); bf16 w[h][i][j] (j contiguous = K)
// ---------------------------------------------------------------------------
__global__ __launch_bounds__(128)
void k1b_softmax(const float* __restrict__ logits, bf16_t* __restrict__ w_bf) {
  const int t = threadIdx.x;
  const size_t base = (size_t)blockIdx.x * NT;  // blockIdx = h*NT + i
  __shared__ float red[128];
  float v[3];
  float mx = -1e30f;
#pragma unroll
  for (int q = 0; q < 3; ++q) { v[q] = logits[base + t + q * 128]; mx = fmaxf(mx, v[q]); }
  red[t] = mx;
  __syncthreads();
  if (t == 0) { float m = red[0]; for (int k = 1; k < 128; ++k) m = fmaxf(m, red[k]); red[0] = m; }
  __syncthreads();
  mx = red[0];
  __syncthreads();
  float sm = 0.f;
#pragma unroll
  for (int q = 0; q < 3; ++q) { v[q] = expf(v[q] - mx); sm += v[q]; }
  red[t] = sm;
  __syncthreads();
  if (t == 0) { float a = 0; for (int k = 0; k < 128; ++k) a += red[k]; red[0] = a; }
  __syncthreads();
  float inv = 1.f / red[0];
#pragma unroll
  for (int q = 0; q < 3; ++q) w_bf[base + t + q * 128] = f2bf(v[q] * inv);
}

// ---------------------------------------------------------------------------
// K2: 32-token tile per block. LN(m), v = mn@wm, g = sigmoid(mn@wg).
// v stored transposed: v_t[s][o=h*32+d][j]  (K=j contiguous, uint4 writes)
// g stored: g[s][i][256] (coalesced)
// ---------------------------------------------------------------------------
__global__ __launch_bounds__(256)
void k2_mpre(const float* __restrict__ m0, const float* __restrict__ nw,
             const float* __restrict__ nb, const float* __restrict__ wm,
             const float* __restrict__ wg, bf16_t* __restrict__ v_t,
             bf16_t* __restrict__ g_bf) {
  const int t = threadIdx.x;
  const int s = blockIdx.x / (NT / 32), i0 = (blockIdx.x % (NT / 32)) * 32;
  __shared__ float xt[32 * CM];
  __shared__ float mn[32 * CM];
  __shared__ float mu_s[32], inv_s[32];
  __shared__ __align__(16) bf16_t vstage[HD * 32];   // [o][ii], 16 KB
  {
    const float4* src = reinterpret_cast<const float4*>(m0 + ((size_t)s * NT + i0) * CM);
    float4* dst = reinterpret_cast<float4*>(xt);
    for (int q = t; q < 512; q += 256) dst[q] = src[q];
  }
  __syncthreads();
  if (t < 32) {
    float s1 = 0.f, s2 = 0.f;
    for (int k = 0; k < CM; ++k) { float v = xt[t * CM + k]; s1 += v; s2 += v * v; }
    float mu = s1 * (1.f / CM);
    mu_s[t]  = mu;
    inv_s[t] = rsqrtf(fmaxf(s2 * (1.f / CM) - mu * mu, 0.f) + EPS);
  }
  __syncthreads();
  for (int q = t; q < 32 * CM; q += 256) {
    int ii = q >> 6, k = q & 63;
    mn[q] = (xt[q] - mu_s[ii]) * inv_s[ii] * nw[k] + nb[k];
  }
  __syncthreads();
  for (int ii = 0; ii < 32; ++ii) {    // thread t owns output column o = t
    float v = 0.f, g = 0.f;
    for (int k = 0; k < CM; ++k) {
      float mk = mn[ii * CM + k];
      v = fmaf(mk, wm[k * HD + t], v);
      g = fmaf(mk, wg[k * HD + t], g);
    }
    vstage[t * 32 + ii] = f2bf(v);
    g_bf[((size_t)s * NT + i0 + ii) * HD + t] = f2bf(sigmoidf_(g));
  }
  __syncthreads();
  {  // row o = t of v_t: 32 contiguous bf16 -> 4 uint4
    const uint4* srow = reinterpret_cast<const uint4*>(vstage + t * 32);
    uint4* drow = reinterpret_cast<uint4*>(v_t + ((size_t)s * HD + t) * NT + i0);
#pragma unroll
    for (int q = 0; q < 4; ++q) drow[q] = srow[q];
  }
}

// ---------------------------------------------------------------------------
// K3: pair-weighted averaging. Block = (s, 16-row i-tile); 8 waves = 8 heads.
//   o[16x32] = w_h[16x384] @ v_{s,h}[384x32]        (WMMA, K=384)
//   gate, then (o*g)[16x256] @ wo[256x64] + m0      (WMMA, K=256)
// All fragments are 8x b32 loads (K-pairs contiguous).
// ---------------------------------------------------------------------------
__global__ __launch_bounds__(256)
void k3_attn(const bf16_t* __restrict__ wbf, const bf16_t* __restrict__ v_t,
             const bf16_t* __restrict__ gbf, const bf16_t* __restrict__ wo_t,
             const float* __restrict__ m0, float* __restrict__ m1) {
  const int t = threadIdx.x, lane = t & 31, wv = t >> 5;
  const int bid = blockIdx.x;               // s*(NT/16) + itile
  const int s = bid / (NT / 16), i0 = (bid % (NT / 16)) * 16;
  __shared__ __align__(16) bf16_t og[16 * HD];   // (o*g), row-major 16x256
  const int arow = lane & 15, akb = (lane >> 4) * 16;
  const int bcol = lane & 15, bkb = (lane >> 4) * 16;
  // ---- phase 1: per-wave head ----
  {
    const int h = wv;
    v8f c0 = {}, c1 = {};
    const bf16_t* wrow = wbf + (size_t)h * NT * NT + (size_t)(i0 + arow) * NT;
    const bf16_t* vb0  = v_t + ((size_t)s * HD + h * DH + bcol) * NT;       // d = bcol
    const bf16_t* vb1  = v_t + ((size_t)s * HD + h * DH + 16 + bcol) * NT;  // d = 16+bcol
    for (int j0 = 0; j0 < NT; j0 += 32) {
      v8u au, b0u, b1u;
#pragma unroll
      for (int r = 0; r < 8; ++r) {
        au[r]  = *reinterpret_cast<const unsigned*>(wrow + j0 + a_koff(akb, r));
        int kb = j0 + bkb + 2 * r;
        b0u[r] = *reinterpret_cast<const unsigned*>(vb0 + kb);
        b1u[r] = *reinterpret_cast<const unsigned*>(vb1 + kb);
      }
      v16bf a = __builtin_bit_cast(v16bf, au);
      c0 = wmma_bf16(a, __builtin_bit_cast(v16bf, b0u), c0);
      c1 = wmma_bf16(a, __builtin_bit_cast(v16bf, b1u), c1);
    }
    const int n = lane & 15, mb = (lane >> 4) * 8;
#pragma unroll
    for (int r = 0; r < 8; ++r) {
      int m = mb + r;
      const bf16_t* gr = gbf + ((size_t)s * NT + i0 + m) * HD + h * DH;
      og[m * HD + h * DH + n]      = f2bf(c0[r] * bf2f(gr[n]));
      og[m * HD + h * DH + 16 + n] = f2bf(c1[r] * bf2f(gr[16 + n]));
    }
  }
  __syncthreads();
  // ---- phase 2: output projection + residual (waves 0..3) ----
  if (wv < 4) {
    const int n0 = wv * 16;
    v8f c = {};
    const bf16_t* brow = wo_t + (size_t)(n0 + bcol) * HD;
    for (int k0 = 0; k0 < HD; k0 += 32) {
      v8u au, bu;
#pragma unroll
      for (int r = 0; r < 8; ++r) {
        au[r] = *reinterpret_cast<const unsigned*>(og + arow * HD + k0 + a_koff(akb, r));
        bu[r] = *reinterpret_cast<const unsigned*>(brow + k0 + bkb + 2 * r);
      }
      c = wmma_bf16(__builtin_bit_cast(v16bf, au), __builtin_bit_cast(v16bf, bu), c);
    }
    const int n = lane & 15, mb = (lane >> 4) * 8;
#pragma unroll
    for (int r = 0; r < 8; ++r) {
      size_t idx = ((size_t)s * NT + i0 + mb + r) * CM + n0 + n;
      m1[idx] = m0[idx] + c[r];
    }
  }
}

// ---------------------------------------------------------------------------
// K4: SwiGLU transition (+residual in place on d_out m-region), fused OPM
// layernorm and a/b projections -> bf16 a_raw/b_raw [i][s][c] (coalesced)
// ---------------------------------------------------------------------------
__global__ __launch_bounds__(256)
void k4_transition(float* __restrict__ m_io,
                   const float* __restrict__ trw, const float* __restrict__ trb,
                   const float* __restrict__ w1, const float* __restrict__ w2,
                   const float* __restrict__ w3,
                   const float* __restrict__ onw, const float* __restrict__ onb,
                   const float* __restrict__ wa, const float* __restrict__ wbm,
                   bf16_t* __restrict__ a_raw, bf16_t* __restrict__ b_raw) {
  const int t = threadIdx.x;
  const size_t row = blockIdx.x;     // s*NT + i
  const int s = (int)(row / NT), i = (int)(row % NT);
  __shared__ float xr[CM], tn[CM], hv[HD], onr[CM], red[CM], stats[2];
  float* io = m_io + row * CM;
  float xv = 0.f;
  if (t < CM) { xv = io[t]; red[t] = xv; }
  __syncthreads();
  if (t == 0) { float a = 0; for (int k = 0; k < CM; ++k) a += red[k]; stats[0] = a * (1.f / CM); }
  __syncthreads();
  if (t < CM) { float d = xv - stats[0]; red[t] = d * d; }
  __syncthreads();
  if (t == 0) { float a = 0; for (int k = 0; k < CM; ++k) a += red[k]; stats[1] = rsqrtf(a * (1.f / CM) + EPS); }
  __syncthreads();
  if (t < CM) { xr[t] = xv; tn[t] = (xv - stats[0]) * stats[1] * trw[t] + trb[t]; }
  __syncthreads();
  {
    float h1 = 0.f, h2 = 0.f;
    for (int k = 0; k < CM; ++k) {
      float tk = tn[k];
      h1 = fmaf(tk, w1[k * HD + t], h1);
      h2 = fmaf(tk, w2[k * HD + t], h2);
    }
    hv[t] = h1 * sigmoidf_(h1) * h2;     // silu(h1)*h2
  }
  __syncthreads();
  float m2v = 0.f;
  if (t < CM) {
    float o = 0.f;
    for (int k = 0; k < HD; ++k) o = fmaf(hv[k], w3[k * CM + t], o);
    m2v = xr[t] + o;
    io[t] = m2v;                         // m2 (final m output)
    red[t] = m2v;
  }
  __syncthreads();
  if (t == 0) { float a = 0; for (int k = 0; k < CM; ++k) a += red[k]; stats[0] = a * (1.f / CM); }
  __syncthreads();
  if (t < CM) { float d = m2v - stats[0]; red[t] = d * d; }
  __syncthreads();
  if (t == 0) { float a = 0; for (int k = 0; k < CM; ++k) a += red[k]; stats[1] = rsqrtf(a * (1.f / CM) + EPS); }
  __syncthreads();
  if (t < CM) onr[t] = (m2v - stats[0]) * stats[1] * onw[t] + onb[t];
  __syncthreads();
  if (t < CM) {
    const int c = t & (CP - 1);
    const float* W = (t < CP) ? wa : wbm;
    float v = 0.f;
    for (int k = 0; k < CM; ++k) v = fmaf(onr[k], W[k * CP + c], v);
    bf16_t hb = f2bf(v);
    size_t idx = ((size_t)i * S + s) * CP + c;
    if (t < CP) a_raw[idx] = hb; else b_raw[idx] = hb;
  }
}

// ---------------------------------------------------------------------------
// K4b: transpose a/b to K-contiguous layout: [i][s][c] -> [i][c][s]
// ---------------------------------------------------------------------------
__global__ __launch_bounds__(256)
void k4b_transpose(const bf16_t* __restrict__ a_raw, const bf16_t* __restrict__ b_raw,
                   bf16_t* __restrict__ a_t, bf16_t* __restrict__ b_t) {
  const int t = threadIdx.x;
  const int i = blockIdx.x / (S / 32), s0 = (blockIdx.x % (S / 32)) * 32;
  __shared__ __align__(16) bf16_t tile[32 * CP];
  for (int pass = 0; pass < 2; ++pass) {
    const bf16_t* src = pass ? b_raw : a_raw;
    bf16_t* dst = pass ? b_t : a_t;
    __syncthreads();
    for (int q = t; q < 128; q += 256)          // 32x32 bf16 = 128 uint4, contiguous
      reinterpret_cast<uint4*>(tile)[q] =
          reinterpret_cast<const uint4*>(src + ((size_t)i * S + s0) * CP)[q];
    __syncthreads();
    for (int q = t; q < 512; q += 256) {        // write rows [c][ss] as dwords
      int c = q >> 4, dw = q & 15;
      unsigned lo = __builtin_bit_cast(unsigned short, tile[(2 * dw) * CP + c]);
      unsigned hi = __builtin_bit_cast(unsigned short, tile[(2 * dw + 1) * CP + c]);
      reinterpret_cast<unsigned*>(dst + ((size_t)i * CP + c) * S + s0)[dw] = lo | (hi << 16);
    }
  }
}

// ---------------------------------------------------------------------------
// K5: outer-product mean + wo projection + z residual.
// Block = 4x4 (i,j) tile; 8 waves x 2 pairs. op[32x32] = a_i^T b_j over S=512
// (WMMA, LDS-staged [ii][c][ss] / [jj][d][ss], uint4 staging + prefetch),
// then fused 16x1024 @ 1024x128 projection (WMMA) and z1 = z0 + zu + bo.
// ---------------------------------------------------------------------------
__global__ __launch_bounds__(256)
void k5_opm(const bf16_t* __restrict__ a_t, const bf16_t* __restrict__ b_t,
            const bf16_t* __restrict__ wo_t, const float* __restrict__ bo,
            const float* __restrict__ z0, float* __restrict__ z1) {
  const int t = threadIdx.x, lane = t & 31, wv = t >> 5;
  const int bi = blockIdx.x / (NT / 4), bj = blockIdx.x % (NT / 4);
  const int i0 = bi * 4, j0 = bj * 4;
  __shared__ __align__(16) bf16_t lds_a[4 * CP * 32];  // [ii][c][ss]
  __shared__ __align__(16) bf16_t lds_b[4 * CP * 32];  // [jj][d][ss]
  __shared__ __align__(16) bf16_t opt[16 * CP * CP];   // 16 pairs x 1024
  const int arow = lane & 15, akb = (lane >> 4) * 16;
  const int bcol = lane & 15, bkb = (lane >> 4) * 16;
  const int p0 = wv * 2;
  v8f acc[2][2][2] = {};
  for (int s0 = 0; s0 < S; s0 += 32) {
    __syncthreads();
    for (int q = t; q < 512; q += 256) {     // [ii(4)][c(32)][r4(4)] uint4
      int ii = q >> 7, c = (q >> 2) & 31, r4 = q & 3;
      reinterpret_cast<uint4*>(lds_a)[q] =
          *reinterpret_cast<const uint4*>(a_t + ((size_t)(i0 + ii) * CP + c) * S + s0 + r4 * 8);
      reinterpret_cast<uint4*>(lds_b)[q] =
          *reinterpret_cast<const uint4*>(b_t + ((size_t)(j0 + ii) * CP + c) * S + s0 + r4 * 8);
    }
    __syncthreads();
    if (s0 + 32 < S) {
      __builtin_prefetch(a_t + ((size_t)i0 * CP) * S + s0 + 32, 0, 1);
      __builtin_prefetch(b_t + ((size_t)j0 * CP) * S + s0 + 32, 0, 1);
    }
#pragma unroll
    for (int pp = 0; pp < 2; ++pp) {
      const int p = p0 + pp;
      const bf16_t* at = lds_a + (p >> 2) * (CP * 32);
      const bf16_t* bt = lds_b + (p & 3) * (CP * 32);
      v16bf afr[2], bfr[2];
#pragma unroll
      for (int hlf = 0; hlf < 2; ++hlf) {
        v8u au, bu;
#pragma unroll
        for (int r = 0; r < 8; ++r) {
          au[r] = *reinterpret_cast<const unsigned*>(at + (hlf * 16 + arow) * 32 + a_koff(akb, r));
          bu[r] = *reinterpret_cast<const unsigned*>(bt + (hlf * 16 + bcol) * 32 + bkb + 2 * r);
        }
        afr[hlf] = __builtin_bit_cast(v16bf, au);
        bfr[hlf] = __builtin_bit_cast(v16bf, bu);
      }
#pragma unroll
      for (int ci = 0; ci < 2; ++ci)
#pragma unroll
        for (int di = 0; di < 2; ++di)
          acc[pp][ci][di] = wmma_bf16(afr[ci], bfr[di], acc[pp][ci][di]);
    }
  }
  __syncthreads();
  {  // op / S -> bf16 rows (16 x 1024) for the projection A-matrix
    const int n = lane & 15, mb = (lane >> 4) * 8;
#pragma unroll
    for (int pp = 0; pp < 2; ++pp) {
      const int p = p0 + pp;
#pragma unroll
      for (int ci = 0; ci < 2; ++ci)
#pragma unroll
        for (int di = 0; di < 2; ++di)
#pragma unroll
          for (int r = 0; r < 8; ++r) {
            int c = ci * 16 + mb + r, d = di * 16 + n;
            opt[p * (CP * CP) + c * CP + d] = f2bf(acc[pp][ci][di][r] * (1.f / (float)S));
          }
    }
  }
  __syncthreads();
  {  // 16 x 1024 @ 1024 x 128; wave wv owns output cols [wv*16, wv*16+16)
    const int n0 = wv * 16;
    v8f c = {};
    const bf16_t* brow = wo_t + (size_t)(n0 + bcol) * (CP * CP);
    for (int k0 = 0; k0 < CP * CP; k0 += 32) {
      v8u au, bu;
#pragma unroll
      for (int r = 0; r < 8; ++r) {
        au[r] = *reinterpret_cast<const unsigned*>(opt + arow * (CP * CP) + k0 + a_koff(akb, r));
        bu[r] = *reinterpret_cast<const unsigned*>(brow + k0 + bkb + 2 * r);
      }
      c = wmma_bf16(__builtin_bit_cast(v16bf, au), __builtin_bit_cast(v16bf, bu), c);
    }
    const int n = lane & 15, mb = (lane >> 4) * 8;
#pragma unroll
    for (int r = 0; r < 8; ++r) {
      int p = mb + r;
      int i = i0 + (p >> 2), j = j0 + (p & 3);
      size_t zi = ((size_t)i * NT + j) * CZ + n0 + n;
      z1[zi] = z0[zi] + c[r] + bo[n0 + n];
    }
  }
}

// ---------------------------------------------------------------------------
extern "C" void kernel_launch(void* const* d_in, const int* in_sizes, int n_in,
                              void* d_out, int out_size, void* d_ws, size_t ws_size,
                              hipStream_t stream) {
  (void)in_sizes; (void)n_in; (void)out_size; (void)ws_size;
  const float* m0      = (const float*)d_in[0];
  const float* z0      = (const float*)d_in[1];
  const float* pwa_nmw = (const float*)d_in[2];
  const float* pwa_nmb = (const float*)d_in[3];
  const float* pwa_nzw = (const float*)d_in[4];
  const float* pwa_nzb = (const float*)d_in[5];
  const float* pwa_wm  = (const float*)d_in[6];
  const float* pwa_wg  = (const float*)d_in[7];
  const float* pwa_wz  = (const float*)d_in[8];
  const float* pwa_wo  = (const float*)d_in[9];
  const float* tr_nw   = (const float*)d_in[10];
  const float* tr_nb   = (const float*)d_in[11];
  const float* tr_w1   = (const float*)d_in[12];
  const float* tr_w2   = (const float*)d_in[13];
  const float* tr_w3   = (const float*)d_in[14];
  const float* opm_nw  = (const float*)d_in[15];
  const float* opm_nb  = (const float*)d_in[16];
  const float* opm_wa  = (const float*)d_in[17];
  const float* opm_wb  = (const float*)d_in[18];
  const float* opm_wo  = (const float*)d_in[19];
  const float* opm_bo  = (const float*)d_in[20];

  auto up256 = [](size_t x) { return (x + 255) & ~(size_t)255; };
  char* p = (char*)d_ws;
  bf16_t* wo_pwa_t = (bf16_t*)p; p += up256((size_t)HD * CM * 2);
  bf16_t* wo_opm_t = (bf16_t*)p; p += up256((size_t)CP * CP * CZ * 2);
  float*  logits   = (float*)p;  p += up256((size_t)H * NT * NT * 4);
  bf16_t* w_bf     = (bf16_t*)p; p += up256((size_t)H * NT * NT * 2);
  bf16_t* v_t      = (bf16_t*)p; p += up256((size_t)S * HD * NT * 2);
  bf16_t* g_bf     = (bf16_t*)p; p += up256((size_t)S * NT * HD * 2);
  bf16_t* a_raw    = (bf16_t*)p; p += up256((size_t)NT * S * CP * 2);
  bf16_t* b_raw    = (bf16_t*)p; p += up256((size_t)NT * S * CP * 2);
  bf16_t* a_t      = (bf16_t*)p; p += up256((size_t)NT * CP * S * 2);
  bf16_t* b_t      = (bf16_t*)p; p += up256((size_t)NT * CP * S * 2);
  // total ws use: ~250 MB

  float* out_m = (float*)d_out;                       // m2: S*NT*CM
  float* out_z = out_m + (size_t)S * NT * CM;         // z1: NT*NT*CZ

  k0_cvt       <<<((CP * CP * CZ) + 255) / 256, 256, 0, stream>>>(pwa_wo, opm_wo, wo_pwa_t, wo_opm_t);
  k1a_bias     <<<(NT * NT + 255) / 256, 256, 0, stream>>>(z0, pwa_nzw, pwa_nzb, pwa_wz, logits);
  k1b_softmax  <<<H * NT, 128, 0, stream>>>(logits, w_bf);
  k2_mpre      <<<S * (NT / 32), 256, 0, stream>>>(m0, pwa_nmw, pwa_nmb, pwa_wm, pwa_wg, v_t, g_bf);
  k3_attn      <<<S * (NT / 16), 256, 0, stream>>>(w_bf, v_t, g_bf, wo_pwa_t, m0, out_m);
  k4_transition<<<S * NT, 256, 0, stream>>>(out_m, tr_nw, tr_nb, tr_w1, tr_w2, tr_w3,
                                            opm_nw, opm_nb, opm_wa, opm_wb, a_raw, b_raw);
  k4b_transpose<<<NT * (S / 32), 256, 0, stream>>>(a_raw, b_raw, a_t, b_t);
  k5_opm       <<<(NT / 4) * (NT / 4), 256, 0, stream>>>(a_t, b_t, wo_opm_t, opm_bo, z0, out_z);
}
